// Critic_2250562863531
// MI455X (gfx1250) — matrix operand order repaired and checked by reading, not compiled, not run on hardware
//
#include <hip/hip_runtime.h>
#include <hip/hip_bf16.h>
#include <stdint.h>

typedef __bf16 bf16_t;
typedef __bf16 v16bf __attribute__((ext_vector_type(16)));
typedef float  v8f   __attribute__((ext_vector_type(8)));

#define TSTEPS 16
#define TPLUS1 17
#define BATCH  2048
#define DDIM   2048
#define HDIM   1024
#define MROWS  (TPLUS1 * BATCH)   // 34816, multiple of 128

#define LROW 72                   // 64 K + 8 pad -> 144B rows, conflict-free b128
#define CROW 136                  // C-tile staging: 128 + 8 pad -> 272B rows (16B aligned)

union FragU { uint4 u[2]; v16bf v; };

// ---- CDNA5 async global->LDS copy (ASYNCcnt path) ---------------------------
#define USE_ASYNC_LDS 1

__device__ __forceinline__ void async_b128(const void* g, void* l) {
#if USE_ASYNC_LDS
  // Low 32 bits of the generic address of a __shared__ object = LDS byte offset.
  uint32_t loff = (uint32_t)(uintptr_t)l;
  asm volatile("global_load_async_to_lds_b128 %0, %1, off"
               :: "v"(loff), "v"(g) : "memory");
#else
  *(uint4*)l = *(const uint4*)g;
#endif
}

__device__ __forceinline__ void wait_async0() {
#if USE_ASYNC_LDS
  asm volatile("s_wait_asynccnt 0x0" ::: "memory");
#endif
}

// ---------------- weight convert + transpose: W[K][N] f32 -> Wt[N][K] bf16 ----
__global__ __launch_bounds__(256) void k_wt_bf16(const float* __restrict__ W,
                                                 bf16_t* __restrict__ Wt,
                                                 int K, int N) {
  __shared__ bf16_t tile[32][33];
  const int k0 = blockIdx.x * 32;
  const int n0 = blockIdx.y * 32;
  const int tid = threadIdx.x;
  for (int i = 0; i < 4; ++i) {
    int slot = i * 256 + tid;
    int r = slot >> 5, c = slot & 31;
    tile[r][c] = (bf16_t)W[(size_t)(k0 + r) * N + (n0 + c)];
  }
  __syncthreads();
  for (int i = 0; i < 4; ++i) {
    int slot = i * 256 + tid;
    int r = slot >> 5, c = slot & 31;
    Wt[(size_t)(n0 + r) * K + (k0 + c)] = tile[c][r];
  }
}

// ---------------- WMMA GEMM + bias + ELU, bf16 out ---------------------------
// C[M,N] = elu(A[M,KDIM] * Bt[N,KDIM]^T + bias), block tile 128x128, BK=64,
// double-buffered LDS (compile-time parity via unroll-2), async global->LDS.
// 8 waves (4 along M x 2 along N); each wave: 2x4 tiles of 16x16.
template <bool A_F32, int KDIM>
__global__ __launch_bounds__(256) void k_gemm_elu(const void* __restrict__ Av,
                                                  const bf16_t* __restrict__ Bt,
                                                  const float* __restrict__ bias,
                                                  bf16_t* __restrict__ Out,
                                                  int M, int N) {
  __shared__ __align__(16) bf16_t As[2][128][LROW];
  __shared__ __align__(16) bf16_t Bs[2][128][LROW];

  const int tid  = threadIdx.x;
  const int lane = tid & 31;
  const int wid  = tid >> 5;
  const int wm   = (wid & 3) * 32;   // wave M offset within block tile
  const int wn   = (wid >> 2) * 64;  // wave N offset within block tile
  const int m0   = blockIdx.x * 128;
  const int n0   = blockIdx.y * 128;

  const float*  Af = (const float*)Av;
  const bf16_t* Ah = (const bf16_t*)Av;

  v8f acc[2][4];
  const v8f vzero = {0.f, 0.f, 0.f, 0.f, 0.f, 0.f, 0.f, 0.f};
  for (int i = 0; i < 2; ++i)
    for (int j = 0; j < 4; ++j) acc[i][j] = vzero;

  const int r  = lane & 15;   // row within 16x16 tile (A) / col (B,C)
  const int hf = lane >> 4;   // lane half selector
  constexpr int NK = KDIM >> 6;

  float4 areg[8];

  // issue one 128x64 bf16 tile, global -> LDS, via async copies (4 b128/thread)
  auto issue_async_tile = [&](const bf16_t* src, bf16_t (*dst)[LROW]) {
    for (int i = 0; i < 4; ++i) {
      int slot = i * 256 + tid;          // 1024 b128 slots
      int row = slot >> 3, c8 = slot & 7;
      async_b128(src + (size_t)row * KDIM + c8 * 8, &dst[row][c8 * 8]);
    }
  };
  // issue 128x64 f32 tile into registers (8 b128/thread, batched)
  auto issue_a_f32 = [&](const float* src) {
    for (int i = 0; i < 8; ++i) {
      int slot = i * 256 + tid;          // 2048 float4 slots
      int row = slot >> 4, c4 = slot & 15;
      areg[i] = *(const float4*)(src + (size_t)row * KDIM + c4 * 4);
    }
  };
  // convert staged f32 regs -> bf16 and store to LDS (uint2 per slot)
  auto store_a_f32 = [&](bf16_t (*dst)[LROW]) {
    for (int i = 0; i < 8; ++i) {
      int slot = i * 256 + tid;
      int row = slot >> 4, c4 = slot & 15;
      union { bf16_t b[4]; uint2 u; } pk;
      pk.b[0] = (bf16_t)areg[i].x; pk.b[1] = (bf16_t)areg[i].y;
      pk.b[2] = (bf16_t)areg[i].z; pk.b[3] = (bf16_t)areg[i].w;
      *(uint2*)&dst[row][c4 * 4] = pk.u;
    }
  };

  // ---- prologue: tile 0 into buffer 0 ----
  issue_async_tile(Bt + (size_t)n0 * KDIM, Bs[0]);
  if (A_F32) {
    issue_a_f32(Af + (size_t)m0 * KDIM);
    store_a_f32(As[0]);
  } else {
    issue_async_tile(Ah + (size_t)m0 * KDIM, As[0]);
  }
  wait_async0();
  __syncthreads();

#pragma unroll 2
  for (int kt = 0; kt < NK; ++kt) {
    const int cur = kt & 1, nxt = cur ^ 1;      // compile-time under unroll-2
    const int ktn = (kt + 1 < NK) ? kt + 1 : kt;  // branch-free; last iter reloads
    const size_t ka = (size_t)ktn << 6;

    // ---- issue next tile (overlaps with WMMA below) ----
    issue_async_tile(Bt + (size_t)n0 * KDIM + ka, Bs[nxt]);
    if (A_F32) issue_a_f32(Af + (size_t)m0 * KDIM + ka);
    else       issue_async_tile(Ah + (size_t)m0 * KDIM + ka, As[nxt]);

    // ---- compute current tile: 2 K-substeps x 8 WMMA ----
    for (int ks = 0; ks < 2; ++ks) {
      FragU fa[2], fb[4];
      // A 16x32 bf16 layout: lane<16 -> row r, K{0..7,16..23}; lane>=16 -> +8
      for (int tm = 0; tm < 2; ++tm) {
        const bf16_t* p = &As[cur][wm + tm * 16 + r][ks * 32 + hf * 8];
        fa[tm].u[0] = *(const uint4*)p;
        fa[tm].u[1] = *(const uint4*)(p + 16);
      }
      // B 32x16 bf16 layout: lane<16 -> col r, K{0..7,16..23}; lane>=16 -> +8
      for (int tn = 0; tn < 4; ++tn) {
        const bf16_t* p = &Bs[cur][wn + tn * 16 + r][ks * 32 + hf * 8];
        fb[tn].u[0] = *(const uint4*)p;
        fb[tn].u[1] = *(const uint4*)(p + 16);
      }
      for (int tm = 0; tm < 2; ++tm)
        for (int tn = 0; tn < 4; ++tn)
          acc[tm][tn] = __builtin_amdgcn_wmma_f32_16x16x32_bf16(
              false, fa[tm].v, false, fb[tn].v, (short)0, acc[tm][tn], false, false);
    }

    // ---- finish next-tile staging, sync buffers ----
    if (A_F32) store_a_f32(As[nxt]);
    wait_async0();
    __syncthreads();
  }

  // ---- epilogue: bias + ELU -> LDS staging -> coalesced b128 stores ----
  // C layout: VGPR e -> M = base + hf*8 + e, N = base + (lane&15)
  bf16_t* cbuf = (bf16_t*)As;   // 128 x CROW staging tile (34 KB <= As size)
  for (int tm = 0; tm < 2; ++tm) {
    const int mrow = wm + tm * 16 + hf * 8;
    for (int tn = 0; tn < 4; ++tn) {
      const int col = wn + tn * 16 + r;
      const float bs = bias[n0 + col];
      for (int e = 0; e < 8; ++e) {
        float v = acc[tm][tn][e] + bs;
        v = v > 0.f ? v : (__expf(v) - 1.f);
        cbuf[(mrow + e) * CROW + col] = (bf16_t)v;
      }
    }
  }
  __syncthreads();
  for (int i = 0; i < 8; ++i) {
    int slot = i * 256 + tid;            // 2048 b128 slots = 128 rows x 16 chunks
    int row = slot >> 4, c8 = slot & 15;
    uint4 v = *(const uint4*)&cbuf[row * CROW + c8 * 8];
    *(uint4*)(Out + (size_t)(m0 + row) * N + n0 + c8 * 8) = v;
  }
}

// ---------------- value head GEMV: value[m] = h[m,:]·Wo + bo -----------------
__global__ __launch_bounds__(256) void k_gemv(const bf16_t* __restrict__ h,
                                              const float* __restrict__ Wo,
                                              const float* __restrict__ bo,
                                              float* __restrict__ value,
                                              int M, int K) {
  const int lane = threadIdx.x & 31;
  const int wid  = threadIdx.x >> 5;
  const int row  = blockIdx.x * 8 + wid;
  if (row >= M) return;
  const bf16_t* hp = h + (size_t)row * K;
  float sum = 0.f;
  for (int i = 0; i < (K >> 8); ++i) {       // 256 elems per iter, 8 per lane
    const int base = (i * 32 + lane) * 8;
    uint4 v = *(const uint4*)(hp + base);
    const bf16_t* e = (const bf16_t*)&v;
    float4 w0 = *(const float4*)(Wo + base);
    float4 w1 = *(const float4*)(Wo + base + 4);
    sum += (float)e[0] * w0.x + (float)e[1] * w0.y +
           (float)e[2] * w0.z + (float)e[3] * w0.w +
           (float)e[4] * w1.x + (float)e[5] * w1.y +
           (float)e[6] * w1.z + (float)e[7] * w1.w;
  }
  for (int off = 16; off > 0; off >>= 1) sum += __shfl_xor(sum, off, 32);
  if (lane == 0) value[row] = sum + bo[0];
}

// ---------------- GAE reverse scan -------------------------------------------
__global__ __launch_bounds__(256) void k_gae(const float* __restrict__ value,
                                             const float* __restrict__ reward,
                                             const float* __restrict__ cont,
                                             float* __restrict__ ret_out,
                                             float* __restrict__ val_out) {
  const int b = blockIdx.x * blockDim.x + threadIdx.x;
  if (b >= BATCH) return;
  float adv = 0.f;
  for (int t = TSTEPS - 1; t >= 0; --t) {
    const float vt   = value[t * BATCH + b];
    const float vt1  = value[(t + 1) * BATCH + b];
    const float disc = cont[(t + 1) * BATCH + b] * 0.99f;
    const float delta = reward[t * BATCH + b] + disc * vt1 - vt;
    adv = delta + disc * 0.95f * adv;
    ret_out[t * BATCH + b] = adv + vt;
    val_out[t * BATCH + b] = vt;
  }
}

extern "C" void kernel_launch(void* const* d_in, const int* in_sizes, int n_in,
                              void* d_out, int out_size, void* d_ws, size_t ws_size,
                              hipStream_t stream) {
  const float* states = (const float*)d_in[0];
  const float* reward = (const float*)d_in[1];
  const float* cont   = (const float*)d_in[2];
  const float* W0 = (const float*)d_in[3];
  const float* b0 = (const float*)d_in[4];
  const float* W1 = (const float*)d_in[5];
  const float* b1 = (const float*)d_in[6];
  const float* W2 = (const float*)d_in[7];
  const float* b2 = (const float*)d_in[8];
  const float* Wo = (const float*)d_in[9];
  const float* bo = (const float*)d_in[10];

  char* ws = (char*)d_ws;
  size_t off = 0;
  auto alloc = [&](size_t bytes) {
    void* p = ws + off;
    off = (off + bytes + 255) & ~(size_t)255;
    return p;
  };
  bf16_t* W0t = (bf16_t*)alloc((size_t)HDIM * DDIM * 2);
  bf16_t* W1t = (bf16_t*)alloc((size_t)HDIM * HDIM * 2);
  bf16_t* W2t = (bf16_t*)alloc((size_t)HDIM * HDIM * 2);
  bf16_t* h0  = (bf16_t*)alloc((size_t)MROWS * HDIM * 2);
  bf16_t* h1  = (bf16_t*)alloc((size_t)MROWS * HDIM * 2);
  float*  val = (float*) alloc((size_t)MROWS * 4);
  bf16_t* h2  = h0;  // layer-2 output reuses h0 (dead after layer 1)

  k_wt_bf16<<<dim3(DDIM / 32, HDIM / 32), 256, 0, stream>>>(W0, W0t, DDIM, HDIM);
  k_wt_bf16<<<dim3(HDIM / 32, HDIM / 32), 256, 0, stream>>>(W1, W1t, HDIM, HDIM);
  k_wt_bf16<<<dim3(HDIM / 32, HDIM / 32), 256, 0, stream>>>(W2, W2t, HDIM, HDIM);

  k_gemm_elu<true,  DDIM><<<dim3(MROWS / 128, HDIM / 128), 256, 0, stream>>>(
      states, W0t, b0, h0, MROWS, HDIM);
  k_gemm_elu<false, HDIM><<<dim3(MROWS / 128, HDIM / 128), 256, 0, stream>>>(
      h0, W1t, b1, h1, MROWS, HDIM);
  k_gemm_elu<false, HDIM><<<dim3(MROWS / 128, HDIM / 128), 256, 0, stream>>>(
      h1, W2t, b2, h2, MROWS, HDIM);

  k_gemv<<<(MROWS + 7) / 8, 256, 0, stream>>>(h2, Wo, bo, val, MROWS, HDIM);

  float* ret_out = (float*)d_out;
  float* val_out = ret_out + (size_t)TSTEPS * BATCH;
  k_gae<<<(BATCH + 255) / 256, 256, 0, stream>>>(val, reward, cont, ret_out, val_out);
}
